// RNN_52106543235311
// MI455X (gfx1250) — matrix operand order repaired and checked by reading, not compiled, not run on hardware
//
#include <hip/hip_runtime.h>

typedef __attribute__((ext_vector_type(16))) _Float16 v16h;
typedef __attribute__((ext_vector_type(2)))  _Float16 v2h;
typedef __attribute__((ext_vector_type(8)))  float    v8f;

#define Bsz 2048
#define Tt  512
#define Ff  64
#define Hh  32

// fused f16 weight layout in workspace (half indices), all [N][K] transposed
#define OFF_MAIN 0                    // [160][128]
#define OFF_GH   (160*128)            // [32][64]
#define OFF_OX   (OFF_GH + 32*64)     // [64][32]
#define OFF_OY   (OFF_OX + 64*32)     // [16][32]
#define TOT_H    (OFF_OY + 16*32)     // 25088 halves = 50176 bytes

// staged-K permutation: stored position c <-> original index perm(c).
// Staged A tiles (h, f_h, f_x) are written interleaved: pos 2n = col n,
// pos 2n+1 = col 16+n, so packed (colA,colB) pairs become one ds_store_b32.
__host__ __device__ __forceinline__ int permK(int c) {
  return (c & 1) ? (16 + (c >> 1)) : (c >> 1);
}

// ---------- prep kernel: build fused f16 weights ----------
__global__ __launch_bounds__(256) void prep_weights(
    const float* __restrict__ W1,  const float* __restrict__ b1,
    const float* __restrict__ Wih, const float* __restrict__ bih,
    const float* __restrict__ Whh, const float* __restrict__ bhh,
    const float* __restrict__ Woh, const float* __restrict__ boh,
    const float* __restrict__ Woy,
    const float* __restrict__ Whx, const float* __restrict__ bhx,
    const float* __restrict__ Wox,
    _Float16* __restrict__ ws)
{
  for (int idx = threadIdx.x; idx < TOT_H; idx += blockDim.x) {
    float v = 0.0f;
    if (idx < OFF_GH) {
      int n = idx >> 7, k = idx & 127;
      if (n < 96) {                       // GRU gate columns (r,z,n)
        int g = n;
        if (k < 64) {                     // x rows: W1 @ Wih_xemb^T
          float s = 0.f;
          for (int e = 0; e < 32; ++e) s += W1[k*32 + e] * Wih[g*65 + e];
          v = s;
        } else if (k < 96) {              // h rows (perm); fold W_hh for r,z
          int j = permK(k - 64);
          v = Wih[g*65 + 32 + j] + ((g < 64) ? Whh[g*32 + j] : 0.f);
        } else if (k == 96) {             // treatment row
          v = Wih[g*65 + 64];
        } else if (k == 98) {             // bias row (A has 1.0 at k=98)
          float s = bih[g] + ((g < 64) ? bhh[g] : 0.f);
          for (int e = 0; e < 32; ++e) s += b1[e] * Wih[g*65 + e];
          v = s;
        }
      } else {                            // U (f_h preact) / V (f_x preact)
        int o = (n < 128) ? (n - 96) : (n - 128);
        const float* Wm = (n < 128) ? Woh : Whx;
        const float* bm = (n < 128) ? boh : bhx;
        if (k < 64) {
          float s = 0.f;
          for (int e = 0; e < 32; ++e) s += W1[k*32 + e] * Wm[e*32 + o];
          v = s;
        } else if (k < 96) {
          int j = permK(k - 64);
          v = Wm[(32 + j)*32 + o];
        } else if (k == 96) {             // treatment (f_hidden idx 65)
          v = Wm[65*32 + o];
        } else if (k == 97) {             // demo_emb (f_hidden idx 64)
          v = Wm[64*32 + o];
        } else if (k == 98) {             // bias
          float s = bm[o];
          for (int e = 0; e < 32; ++e) s += b1[e] * Wm[e*32 + o];
          v = s;
        }
      }
    } else if (idx < OFF_OX) {            // gh_n: [32][64], h rows permuted
      int l = idx - OFF_GH, c = l >> 6, k = l & 63;
      if (k < 32)       v = Whh[(64 + c)*32 + permK(k)];
      else if (k == 34) v = bhh[64 + c];  // bias via A's 1.0 at local k=34
    } else if (idx < OFF_OY) {            // W_ox^T: [64][32], rows permuted
      int l = idx - OFF_OX, c = l >> 5, k = l & 31;
      v = Wox[permK(k)*64 + c];
    } else {                              // W_oy padded: [16][32], col 0 only
      int l = idx - OFF_OY, c = l >> 5, k = l & 31;
      v = (c == 0) ? Woy[permK(k)] : 0.f;
    }
    ws[idx] = (_Float16)v;
  }
}

// ---------- helpers ----------
__device__ __forceinline__ v8f vzero() {
  v8f z;
#pragma unroll
  for (int i = 0; i < 8; ++i) z[i] = 0.f;
  return z;
}

__device__ __forceinline__ v8f wmma(v16h a, v16h b, v8f c) {
  return __builtin_amdgcn_wmma_f32_16x16x32_f16(false, a, false, b, (short)0, c,
                                                false, false);
}

// B fragment (32x16, K x N) from [N][K]-transposed f16: lane holds col n=lane%16,
// K = kt*32 + (lane/16)*8 + {0..7, 16..23}.
__device__ __forceinline__ v16h ldB(const _Float16* p, int K, int nt, int kt, int lane) {
  const _Float16* q = p + (nt*16 + (lane & 15))*K + kt*32 + ((lane >> 4) << 3);
  v16h o;
#pragma unroll
  for (int i = 0; i < 8; ++i) { o[i] = q[i]; o[i + 8] = q[i + 16]; }
  return o;
}

// A fragment (16x32 f16) from row-major [16][32] f16 tile in LDS
__device__ __forceinline__ v16h ldA16(const _Float16* p, int lane) {
  const _Float16* r = p + ((lane & 15) << 5) + ((lane >> 4) << 3);
  v16h o;
#pragma unroll
  for (int i = 0; i < 8; ++i) { o[i] = r[i]; o[i + 8] = r[i + 16]; }
  return o;
}

// A fragment from row-major f32 (row pointer for this lane's M-row), with cvt
__device__ __forceinline__ v16h ldA32(const float* rowp, int lane) {
  const float* r = rowp + ((lane >> 4) << 3);
  v16h o;
#pragma unroll
  for (int i = 0; i < 8; ++i) {
    o[i]     = (_Float16)r[i];
    o[i + 8] = (_Float16)r[i + 16];
  }
  return o;
}

__device__ __forceinline__ float sigm(float x) {
  return 1.f / (1.f + __expf(-x));
}
__device__ __forceinline__ float tanh_fast(float x) {
  float xc = fminf(fmaxf(x, -20.f), 20.f);
  float e = __expf(-2.f * xc);
  return (1.f - e) / (1.f + e);
}
__device__ __forceinline__ float softplus_f(float x) {
  return fmaxf(x, 0.f) + __logf(1.f + __expf(-fabsf(x)));
}

// ---------- main scan kernel: 1 wave per 16-row batch tile ----------
#define WAVES 4
__global__ __launch_bounds__(WAVES * 32) void rnn_scan(
    const float* __restrict__ x,      // [B, T+1, F]
    const float* __restrict__ xdemo,  // [B, 1]
    const float* __restrict__ tr,     // [B, T]
    const float* __restrict__ yin,    // [B, T+1]
    const float* __restrict__ h0,     // [B, H]
    const float* __restrict__ Wdemo,  // [1,1]
    const float* __restrict__ bdemo,  // [1]
    const _Float16* __restrict__ ws,  // fused weights
    float* __restrict__ out)          // [B, T, 97]
{
  __shared__ alignas(32) _Float16 sW[TOT_H];
  __shared__ alignas(32) _Float16 sStage[WAVES][3][16 * 32];

  for (int i = threadIdx.x; i < TOT_H; i += blockDim.x) sW[i] = ws[i];
  __syncthreads();

  const _Float16* Bmain = sW + OFF_MAIN;
  const _Float16* Bgh   = sW + OFF_GH;
  const _Float16* Box   = sW + OFF_OX;
  const _Float16* Boy   = sW + OFF_OY;

  const int lane = threadIdx.x & 31;
  const int wv   = threadIdx.x >> 5;
  const int tile = blockIdx.x * WAVES + wv;   // 0..127
  const int r0   = tile << 4;
  const int mr   = lane & 15;
  const int kh   = lane >> 4;

  _Float16* stH  = &sStage[wv][0][0];
  _Float16* stFH = &sStage[wv][1][0];
  _Float16* stFX = &sStage[wv][2][0];

  // h init: A-layout f16 fragment (staged-K permuted) + fp32 C/D-layout copy
  const float* rowh = h0 + (size_t)(r0 + mr) * Hh;
  v16h ah;
#pragma unroll
  for (int i = 0; i < 16; ++i) {
    int c = ((i < 8) ? i : (i + 8)) + (kh << 3);   // tile-local staged pos
    ah[i] = (_Float16)rowh[permK(c)];
  }
  v8f hA, hB;
#pragma unroll
  for (int v = 0; v < 8; ++v) {
    int m = v + (kh << 3);
    hA[v] = h0[(size_t)(r0 + m) * Hh + mr];
    hB[v] = h0[(size_t)(r0 + m) * Hh + 16 + mr];
  }

  // aux A-tile (k: 0=tr, 1=demo_emb, 2=1.0, rest 0). Lanes 16-31 map to
  // all-zero B rows, so no lane predicate needed (their values are don't-care).
  float dem = xdemo[r0 + mr] * Wdemo[0] + bdemo[0];
  v16h aux;
#pragma unroll
  for (int i = 0; i < 16; ++i) aux[i] = (_Float16)0.f;
  aux[1] = (_Float16)dem;
  aux[2] = (_Float16)1.0f;

  for (int t = 0; t < Tt; ++t) {
    aux[0] = (_Float16)tr[(size_t)(r0 + mr) * Tt + t];

    const float* xr = x + ((size_t)(r0 + mr) * (Tt + 1) + t) * Ff;
    // prefetch next timestep's x row (global_prefetch_b8) to hide latency
    __builtin_prefetch(xr + Ff, 0, 3);
    __builtin_prefetch(xr + Ff + 32, 0, 3);

    v16h ax0 = ldA32(xr, lane);        // x features 0..31
    v16h ax1 = ldA32(xr + 32, lane);   // x features 32..63

    // fused GEMM: A[16x128] @ B[128x160] -> r(0,1) z(2,3) gi_n(4,5) U(6,7) V(8,9)
    v8f acc[10];
#pragma unroll
    for (int nt = 0; nt < 10; ++nt) {
      v8f c = vzero();
      c = wmma(ax0, ldB(Bmain, 128, nt, 0, lane), c);
      c = wmma(ax1, ldB(Bmain, 128, nt, 1, lane), c);
      c = wmma(ah,  ldB(Bmain, 128, nt, 2, lane), c);
      c = wmma(aux, ldB(Bmain, 128, nt, 3, lane), c);
      acc[nt] = c;
    }
    // gh_n = h @ W_hh[2H:3H]^T + b_hh_n  (A tiles: h, aux)
    v8f ghA = wmma(aux, ldB(Bgh, 64, 0, 1, lane),
                   wmma(ah, ldB(Bgh, 64, 0, 0, lane), vzero()));
    v8f ghB = wmma(aux, ldB(Bgh, 64, 1, 1, lane),
                   wmma(ah, ldB(Bgh, 64, 1, 0, lane), vzero()));

    // gates + heads (elementwise, fp32); stage f16 tiles to LDS as packed
    // (colA,colB) pairs -> single ds_store_b32 each (interleaved K order)
#pragma unroll
    for (int v = 0; v < 8; ++v) {
      float rA = sigm(acc[0][v]), rB = sigm(acc[1][v]);
      float zA = sigm(acc[2][v]), zB = sigm(acc[3][v]);
      float nA = tanh_fast(acc[4][v] + rA * ghA[v]);
      float nB = tanh_fast(acc[5][v] + rB * ghB[v]);
      float hAn = (1.f - zA) * nA + zA * hA[v];
      float hBn = (1.f - zB) * nB + zB * hB[v];
      hA[v] = hAn; hB[v] = hBn;
      float fhA = fmaxf(acc[6][v], 0.f), fhB = fmaxf(acc[7][v], 0.f);
      float fxA = softplus_f(acc[8][v]), fxB = softplus_f(acc[9][v]);
      int m = v + (kh << 3);
      int o2 = (m << 5) + (mr << 1);
      v2h p;
      p[0] = (_Float16)hAn; p[1] = (_Float16)hBn;
      *(v2h*)(stH + o2) = p;
      p[0] = (_Float16)fhA; p[1] = (_Float16)fhB;
      *(v2h*)(stFH + o2) = p;
      p[0] = (_Float16)fxA; p[1] = (_Float16)fxB;
      *(v2h*)(stFX + o2) = p;
    }
    v16h afh = ldA16(stFH, lane);
    v16h afx = ldA16(stFX, lane);
    ah = ldA16(stH, lane);             // h_new A-fragment for next step

    // x_new = f_x @ W_ox  (K=32, N=64)
    v8f xo[4];
#pragma unroll
    for (int j = 0; j < 4; ++j)
      xo[j] = wmma(afx, ldB(Box, 32, j, 0, lane), vzero());
    // y = f_h @ W_oy (padded N=16, col 0 valid)
    v8f yo = wmma(afh, ldB(Boy, 32, 0, 0, lane), vzero());

    // write out[b, t, 0:97] = [y, x_new(64), h_new(32)]
#pragma unroll
    for (int v = 0; v < 8; ++v) {
      int m = v + (kh << 3);
      size_t ob = ((size_t)(r0 + m) * Tt + t) * 97;
#pragma unroll
      for (int j = 0; j < 4; ++j) out[ob + 1 + j * 16 + mr] = xo[j][v];
      out[ob + 65 + mr] = hA[v];
      out[ob + 81 + mr] = hB[v];
      if (mr == 0) {
        float yv = yo[v];
        if (t > 0) yv += yin[(size_t)(r0 + m) * (Tt + 1) + t];
        out[ob] = yv;
      }
    }
  }
}

// ---------- launch ----------
extern "C" void kernel_launch(void* const* d_in, const int* in_sizes, int n_in,
                              void* d_out, int out_size, void* d_ws, size_t ws_size,
                              hipStream_t stream) {
  const float* x      = (const float*)d_in[0];
  const float* xdemo  = (const float*)d_in[1];
  const float* ftr    = (const float*)d_in[2];
  const float* yin    = (const float*)d_in[3];
  const float* h0     = (const float*)d_in[4];
  const float* W1     = (const float*)d_in[5];
  const float* b1     = (const float*)d_in[6];
  const float* Wdemo  = (const float*)d_in[7];
  const float* bdemo  = (const float*)d_in[8];
  const float* Wih    = (const float*)d_in[9];
  const float* bih    = (const float*)d_in[10];
  const float* Whh    = (const float*)d_in[11];
  const float* bhh    = (const float*)d_in[12];
  const float* Woh    = (const float*)d_in[13];
  const float* boh    = (const float*)d_in[14];
  const float* Woy    = (const float*)d_in[15];
  const float* Whx    = (const float*)d_in[16];
  const float* bhx    = (const float*)d_in[17];
  const float* Wox    = (const float*)d_in[18];

  _Float16* ws = (_Float16*)d_ws;
  float* out = (float*)d_out;

  prep_weights<<<1, 256, 0, stream>>>(W1, b1, Wih, bih, Whh, bhh,
                                      Woh, boh, Woy, Whx, bhx, Wox, ws);
  rnn_scan<<<Bsz / (16 * WAVES), WAVES * 32, 0, stream>>>(
      x, xdemo, ftr, yin, h0, Wdemo, bdemo, ws, out);
}
